// CFCNConstructor_5446018532063
// MI455X (gfx1250) — matrix operand order repaired and checked by reading, not compiled, not run on hardware
//
#include <hip/hip_runtime.h>

// ---------------------------------------------------------------------------
// Per-component MLP (D=64 tiny MLPs), fused, transposed-GEMM form for gfx1250.
//   h1 = relu(x[:,d,None] * W1[d] + b1[d])            (built in regs, B operand)
//   h2T = W2[d]^T @ h1^T   (+b2, relu)                (bf16 WMMA, f32 acc)
//   out[:,d] = sum_h h2 * W3[d] + b3[d]               (in-lane + shfl_xor(16))
//
// A (=W2^T, invariant) streams from LDS in fragment-major layout: per-lane 8
// contiguous dwords -> two ds_load_b128 straight into the WMMA source tuple,
// each fragment reused by 4 batch tiles (4 WMMAs per fragment load).
//
// The mm loop is kept rolled and fenced with sched_barrier(0) so only ONE
// mm-group of accumulators (32 VGPRs) is live at a time -- rounds 2/3 showed
// the scheduler otherwise interleaves all groups, forcing VGPR-MSB mode and
// scratch spills.
// ---------------------------------------------------------------------------

typedef __attribute__((ext_vector_type(16))) __bf16 v16bf;
typedef __attribute__((ext_vector_type(8)))  float  v8f;

#define B_ROWS   16384
#define D_COMP   64
#define H_DIM    128
#define SPLITS   16                       // blocks per component
#define ROWS_PER_BLOCK (B_ROWS / SPLITS)  // 1024
#define NTILES   4                        // batch tiles (16 rows) per wave
#define ROWS_PER_PASS (8 * NTILES * 16)   // 512 rows per block pass
#define ALS      12                       // A-frag lane stride in dwords (48B, 16B-aligned)
#define AFRAG_DW (32 * ALS)               // dwords per staged fragment

__global__ __launch_bounds__(256)
void cfcn_fused_kernel(const float* __restrict__ x,
                       const float* __restrict__ W1,
                       const float* __restrict__ b1,
                       const float* __restrict__ W2,
                       const float* __restrict__ b2,
                       const float* __restrict__ W3,
                       const float* __restrict__ b3,
                       float* __restrict__ out)
{
    // 32 A-fragments (8 mm x 4 kk) of W2^T in bf16, fragment-major: 48 KB
    __shared__ __align__(16) unsigned ldsA[32 * AFRAG_DW];
    __shared__ __align__(16) float ldsW1[H_DIM], ldsB1[H_DIM];
    __shared__ __align__(16) float ldsB2[H_DIM], ldsW3[H_DIM];

    const int d     = blockIdx.x / SPLITS;
    const int chunk = blockIdx.x % SPLITS;
    const int tid   = threadIdx.x;
    const int wave  = tid >> 5;
    const int lane  = tid & 31;
    const int lhalf = lane >> 4;      // 0: lanes 0-15, 1: lanes 16-31
    const int lmod  = lane & 15;

    // ---- stage per-component vectors --------------------------------------
    if (tid < H_DIM) {
        ldsW1[tid] = W1[d * H_DIM + tid];
        ldsB1[tid] = b1[d * H_DIM + tid];
        ldsB2[tid] = b2[d * H_DIM + tid];
        ldsW3[tid] = W3[d * H_DIM + tid];
    }

    // ---- stage W2^T as bf16 A-fragments (fragment-major) ------------------
    // A-layout (16-bit, 16x32): lane l: M = mm*16 + l%16,
    //   K = kk*32 + (l>=16?8:0) + 2v + (v>=4?8:0), pair {K, K+1} packed per dword.
    // Dword for (frag f, lane l, v) at ldsA[f*AFRAG_DW + l*ALS + v].
    {
        const float* W2d = W2 + (size_t)d * H_DIM * H_DIM;
        const int l  = tid & 31;
        const int vv = tid >> 5;                 // 0..7
        const int lm = l & 15, lh = l >> 4;
        #pragma unroll 1
        for (int f = 0; f < 32; ++f) {           // one fragment per step
            const int mm = f >> 2, kk = f & 3;
            const int M = mm * 16 + lm;
            const int K = kk * 32 + lh * 8 + 2 * vv + (vv >= 4 ? 8 : 0);
            // W2^T[M][K] = W2[d][K][M]  (coalesced along M)
            const float a0 = W2d[K * H_DIM + M];
            const float a1 = W2d[(K + 1) * H_DIM + M];
            union { __bf16 h[2]; unsigned u; } p;
            p.h[0] = (__bf16)a0;
            p.h[1] = (__bf16)a1;
            ldsA[f * AFRAG_DW + l * ALS + vv] = p.u;
        }
    }
    __syncthreads();

    const float b3v = b3[d];

    #pragma unroll 1
    for (int it = 0; it < ROWS_PER_BLOCK / ROWS_PER_PASS; ++it) {
        const int waveBase = chunk * ROWS_PER_BLOCK + it * ROWS_PER_PASS
                           + wave * (NTILES * 16);

        // Each lane owns batch row N = lmod of each of its 4 tiles.
        float xv[NTILES];
        #pragma unroll
        for (int t = 0; t < NTILES; ++t)
            xv[t] = x[(size_t)(waveBase + t * 16 + lmod) * D_COMP + d];

        // ---- build B fragments (h1^T), 16-bit B layout --------------------
        // lane l: N = l%16, K = kk*32 + (l>=16?8:0) + 2v + (v>=4?8:0) + p
        // (W1,b1) pair per (kk,v) shared by all 4 tiles.
        v16bf bfr[NTILES][4];
        #pragma unroll
        for (int kk = 0; kk < 4; ++kk) {
            #pragma unroll
            for (int v = 0; v < 8; ++v) {
                const int k0 = kk * 32 + lhalf * 8 + 2 * v + (v >= 4 ? 8 : 0);   // even -> 8B aligned
                const float2 wv = *(const float2*)&ldsW1[k0];
                const float2 bv = *(const float2*)&ldsB1[k0];
                #pragma unroll
                for (int t = 0; t < NTILES; ++t) {
                    const float f0 = fmaxf(xv[t] * wv.x + bv.x, 0.0f);
                    const float f1 = fmaxf(xv[t] * wv.y + bv.y, 0.0f);
                    bfr[t][kk][2 * v]     = (__bf16)f0;
                    bfr[t][kk][2 * v + 1] = (__bf16)f1;
                }
            }
        }
        __builtin_amdgcn_sched_barrier(0);   // fence: B-build fully before GEMM

        float part[NTILES];
        #pragma unroll
        for (int t = 0; t < NTILES; ++t) part[t] = 0.0f;

        // ---- GEMM: h2T tile (mm) x batch tiles (t), A streamed from LDS ---
        // Rolled + fenced: exactly one mm-group of accumulators live at a time.
        #pragma unroll 1
        for (int mm = 0; mm < 8; ++mm) {
            v8f acc[NTILES] = {};
            #pragma unroll
            for (int kk = 0; kk < 4; ++kk) {
                const v16bf av = *(const v16bf*)
                    &ldsA[(mm * 4 + kk) * AFRAG_DW + lane * ALS];  // 2x ds_load_b128
                #pragma unroll
                for (int t = 0; t < NTILES; ++t)
                    acc[t] = __builtin_amdgcn_wmma_f32_16x16x32_bf16(
                                 false, av, false, bfr[t][kk], (short)0,
                                 acc[t], false, false);
            }
            // C layout: VGPR r -> h_out = mm*16 + 8*lhalf + r, lane -> row lmod.
            const int h0 = mm * 16 + 8 * lhalf;
            const float4 b2lo = *(const float4*)&ldsB2[h0];
            const float4 b2hi = *(const float4*)&ldsB2[h0 + 4];
            const float4 w3lo = *(const float4*)&ldsW3[h0];
            const float4 w3hi = *(const float4*)&ldsW3[h0 + 4];
            const float b2a[8] = {b2lo.x, b2lo.y, b2lo.z, b2lo.w,
                                  b2hi.x, b2hi.y, b2hi.z, b2hi.w};
            const float w3a[8] = {w3lo.x, w3lo.y, w3lo.z, w3lo.w,
                                  w3hi.x, w3hi.y, w3hi.z, w3hi.w};
            #pragma unroll
            for (int t = 0; t < NTILES; ++t) {
                #pragma unroll
                for (int r = 0; r < 8; ++r) {
                    const float h2 = fmaxf(acc[t][r] + b2a[r], 0.0f);
                    part[t] += h2 * w3a[r];
                }
            }
            __builtin_amdgcn_sched_barrier(0);  // no cross-iteration interleave
        }

        // ---- combine lane halves (rows live in lanes l and l+16) ----------
        #pragma unroll
        for (int t = 0; t < NTILES; ++t) {
            const float tot = part[t] + __shfl_xor(part[t], 16, 32);
            if (lane < 16)
                out[(size_t)(waveBase + t * 16 + lane) * D_COMP + d] = tot + b3v;
        }
    }
}

extern "C" void kernel_launch(void* const* d_in, const int* in_sizes, int n_in,
                              void* d_out, int out_size, void* d_ws, size_t ws_size,
                              hipStream_t stream) {
    (void)in_sizes; (void)n_in; (void)out_size; (void)d_ws; (void)ws_size;
    const float* x  = (const float*)d_in[0];
    const float* W1 = (const float*)d_in[1];
    const float* b1 = (const float*)d_in[2];
    const float* W2 = (const float*)d_in[3];
    const float* b2 = (const float*)d_in[4];
    const float* W3 = (const float*)d_in[5];
    const float* b3 = (const float*)d_in[6];
    float* out = (float*)d_out;

    dim3 grid(D_COMP * SPLITS);   // 1024 blocks
    dim3 block(256);              // 8 wave32 waves
    cfcn_fused_kernel<<<grid, block, 0, stream>>>(x, W1, b1, W2, b2, W3, b3, out);
}
